// HyenaFilter_35459249995894
// MI455X (gfx1250) — matrix operand order repaired and checked by reading, not compiled, not run on hardware
//
#include <hip/hip_runtime.h>
#include <stdint.h>

// Problem constants (fixed by the reference).
#define BDIM 4
#define DDIM 768
#define LDIM 8192
#define NT   (LDIM / 16)              // 512 time tiles of 16
#define GROUP_TILES 8                 // output tiles per column group (8 tiles x 2 batches = 16 N cols)
#define NGROUPS (NT / GROUP_TILES)    // 64 column groups per (d, batch-pair)
#define WAVES_PER_WG 16
#define GROUPS_PER_WAVE (NGROUPS / WAVES_PER_WG)  // 4
#define THREADS (WAVES_PER_WG * 32)   // 512 (wave32)
#define HREV_LEN (LDIM + 32)          // reversed filter + zero pad for causal boundary

typedef __attribute__((ext_vector_type(16))) __bf16 v16bf;
typedef __attribute__((ext_vector_type(8)))  float  v8f;
typedef __attribute__((ext_vector_type(4)))  unsigned int v4u;
typedef __attribute__((ext_vector_type(8)))  int    v8i;
typedef __attribute__((ext_vector_type(4)))  int    v4i;

union alignas(32) Frag16 {
  v16bf v;
  unsigned short u[16];
  uint4 q[2];
};

static __device__ inline unsigned short f2bf(float f) {
  union { float f; unsigned u; } c;
  c.f = f;
  unsigned u = c.u + 0x7FFFu + ((c.u >> 16) & 1u);  // round-to-nearest-even
  return (unsigned short)(u >> 16);
}

// A-fragment: Toeplitz gather from the reversed filter in LDS.
// A layout (contiguous K, matching B): lanes 0-15 hold K=0..15; lanes 16-31 hold K=16..31.
#define LOAD_AFRAG(a, kp)                                        \
  {                                                              \
    const int Q_  = (LDIM - 1) - 16 * (kp) - low;                \
    const int c1_ = Q_ - 16 * hi;                                \
    const int c2_ = Q_ - 16 * hi + 8;                            \
    _Pragma("unroll")                                            \
    for (int e_ = 0; e_ < 8; ++e_) (a).u[e_] = hrev[c1_ + e_];   \
    _Pragma("unroll")                                            \
    for (int e_ = 0; e_ < 8; ++e_) (a).u[8 + e_] = hrev[c2_ + e_]; \
  }

// B-fragment: contiguous bf16 x tile (lanes 0-15: K=0..15 of tile I-kp,
// lanes 16-31: K=16..31 = tile I-kp-1), zero for pre-causal tiles.
#define LOAD_BFRAG(xf, I0, kp)                                   \
  {                                                              \
    const int m_ = (I0) + qx - (kp) - hi;                        \
    if (m_ >= 0) {                                               \
      const uint4* p_ = (const uint4*)&xs[blx * LDIM + 16 * m_]; \
      (xf).q[0] = p_[0];                                         \
      (xf).q[1] = p_[1];                                         \
    } else {                                                     \
      (xf).q[0] = make_uint4(0u, 0u, 0u, 0u);                    \
      (xf).q[1] = make_uint4(0u, 0u, 0u, 0u);                    \
    }                                                            \
  }

__global__ __launch_bounds__(THREADS)
void hyena_causal_conv_wmma(const float* __restrict__ x,
                            const float* __restrict__ h,
                            const float* __restrict__ bias,
                            float* __restrict__ out) {
  // 49,216 bytes of static LDS. xs doubles as the f32 TDM landing zone for h
  // (32 KB) before being overwritten with the bf16 x operands.
  __shared__ alignas(16) unsigned short xs[2 * LDIM];
  __shared__ alignas(16) unsigned short hrev[HREV_LEN];

  const int d    = blockIdx.x >> 1;
  const int pair = blockIdx.x & 1;
  const int b0   = pair * 2;
  const int tid  = threadIdx.x;
  const int wave = tid >> 5;
  const int lane = tid & 31;
  const int low  = lane & 15;   // A: row i;  B/D: column index N
  const int hi   = lane >> 4;   // K-half select (lag tile kp vs kp+1)
  const int blx  = low >> 3;    // batch within pair (cols 0-7 -> b0, 8-15 -> b0+1)
  const int qx   = low & 7;     // output tile within group

  // ---- Phase 0: TDM the f32 filter row (32 KB contiguous) into LDS ----
  float* hf = (float*)xs;
  if (wave == 0) {
    const unsigned long long ga =
        (unsigned long long)(const void*)(h + (size_t)d * LDIM);
    const unsigned lds_addr = (unsigned)(size_t)(void*)xs;
    v4u g0;
    g0[0] = 1u;                                   // count=1, user descriptor
    g0[1] = lds_addr;                             // lds_addr [63:32]
    g0[2] = (unsigned)(ga & 0xFFFFFFFFu);         // global_addr lo
    g0[3] = (unsigned)((ga >> 32) & 0x01FFFFFFu)  // global_addr [56:32]
            | (2u << 30);                         // type=2 ("image")
    v8i g1;
    g1[0] = (int)(2u << 16);                      // workgroup_mask=0, data_size=4B
    g1[1] = (int)((LDIM & 0xFFFF) << 16);         // tensor_dim0[15:0] -> bits[63:48]
    g1[2] = (int)(1u << 16) | (int)(LDIM >> 16);  // tensor_dim0 hi, tensor_dim1=1
    g1[3] = (int)(LDIM << 16);                    // tile_dim0=8192 -> bits[127:112]
    g1[4] = 0;                                    // tile_dim1=0, tile_dim2=0
    g1[5] = LDIM;                                 // tensor_dim0_stride lo
    g1[6] = 0;
    g1[7] = 0;
    v4i g2 = {0, 0, 0, 0};
    v4i g3 = {0, 0, 0, 0};
    v8i g4 = {0, 0, 0, 0, 0, 0, 0, 0};
    __builtin_amdgcn_tensor_load_to_lds(g0, g1, g2, g3, g4, 0);
    __builtin_amdgcn_s_wait_tensorcnt(0);
  }
  __syncthreads();

  // ---- Phase 1: f32 filter (LDS) -> reversed bf16 hrev, zero pad ----
  #pragma unroll
  for (int it = 0; it < 16; ++it) {
    int j = tid + THREADS * it;                   // 0..8191
    hrev[(LDIM - 1) - j] = f2bf(hf[j]);
  }
  if (tid < 32) hrev[LDIM + tid] = 0;
  __syncthreads();

  // ---- Phase 2: stage x for this batch pair as bf16 (overwrites hf) ----
  #pragma unroll
  for (int bl = 0; bl < 2; ++bl) {
    const float4* src = (const float4*)(x + ((size_t)(b0 + bl) * DDIM + d) * LDIM);
    __builtin_prefetch(src, 0, 1);
    #pragma unroll
    for (int it = 0; it < 4; ++it) {
      int v = tid + THREADS * it;                 // 0..2047 float4s
      float4 f = src[v];
      int base = bl * LDIM + 4 * v;
      xs[base + 0] = f2bf(f.x);
      xs[base + 1] = f2bf(f.y);
      xs[base + 2] = f2bf(f.z);
      xs[base + 3] = f2bf(f.w);
    }
  }
  __syncthreads();

  // ---- Phase 3: lag loop, 2 lag-tile pairs (K=64) per iteration ----
  v8f zero = {};
  v8f acc[GROUPS_PER_WAVE];
  int I0r[GROUPS_PER_WAVE];
  #pragma unroll
  for (int r = 0; r < GROUPS_PER_WAVE; ++r) {
    acc[r] = zero;
    I0r[r] = GROUP_TILES * (wave + WAVES_PER_WG * r);
  }
  const int kpmax = I0r[GROUPS_PER_WAVE - 1] + GROUP_TILES;  // multiple of 4

  for (int kp = 0; kp < kpmax; kp += 4) {
    Frag16 a0, a1;
    LOAD_AFRAG(a0, kp);
    LOAD_AFRAG(a1, kp + 2);

    #pragma unroll
    for (int r = 0; r < GROUPS_PER_WAVE; ++r) {
      const int I0 = I0r[r];
      if (kp < I0 + GROUP_TILES) {        // wave-uniform scalar branch
        Frag16 xf;
        LOAD_BFRAG(xf, I0, kp);
        acc[r] = __builtin_amdgcn_wmma_f32_16x16x32_bf16(
            false, a0.v, false, xf.v, (short)0, acc[r], false, false);
      }
      if (kp + 2 < I0 + GROUP_TILES) {
        Frag16 xf;
        LOAD_BFRAG(xf, I0, kp + 2);
        acc[r] = __builtin_amdgcn_wmma_f32_16x16x32_bf16(
            false, a1.v, false, xf.v, (short)0, acc[r], false, false);
      }
    }
  }

  // ---- Epilogue: y = conv + b[d]*x (skip term in f32 from original x) ----
  const float bscale = bias[d];
  const size_t rowbase = ((size_t)(b0 + blx) * DDIM + d) * LDIM;
  #pragma unroll
  for (int r = 0; r < GROUPS_PER_WAVE; ++r) {
    const int I  = I0r[r] + qx;
    const int t0 = 16 * I + 8 * hi;       // D layout: M = elem + 8*(lane>=16)
    const float4* xin = (const float4*)(x + rowbase + t0);
    float4 xa = xin[0];
    float4 xb = xin[1];
    float4 o0, o1;
    o0.x = acc[r][0] + bscale * xa.x;
    o0.y = acc[r][1] + bscale * xa.y;
    o0.z = acc[r][2] + bscale * xa.z;
    o0.w = acc[r][3] + bscale * xa.w;
    o1.x = acc[r][4] + bscale * xb.x;
    o1.y = acc[r][5] + bscale * xb.y;
    o1.z = acc[r][6] + bscale * xb.z;
    o1.w = acc[r][7] + bscale * xb.w;
    float4* op = (float4*)(out + rowbase + t0);
    op[0] = o0;
    op[1] = o1;
  }
}

extern "C" void kernel_launch(void* const* d_in, const int* in_sizes, int n_in,
                              void* d_out, int out_size, void* d_ws, size_t ws_size,
                              hipStream_t stream) {
  (void)in_sizes; (void)n_in; (void)out_size; (void)d_ws; (void)ws_size;
  const float* x    = (const float*)d_in[0];
  const float* h    = (const float*)d_in[1];
  const float* bias = (const float*)d_in[2];
  float* out        = (float*)d_out;

  dim3 grid(DDIM * 2);   // one WG per (d, batch-pair)
  dim3 block(THREADS);   // 16 waves (wave32)
  hipLaunchKernelGGL(hyena_causal_conv_wmma, grid, block, 0, stream,
                     x, h, bias, out);
}
